// CausalAttention_25297357374003
// MI455X (gfx1250) — matrix operand order repaired and checked by reading, compiled, not run on hardware
//
#include <hip/hip_runtime.h>

#define B_  4
#define T_  2048
#define D_  768
#define H_  12
#define HD_ 64
#define BT_ (B_ * T_)

typedef __attribute__((ext_vector_type(16))) _Float16 v16h;
typedef __attribute__((ext_vector_type(8)))  _Float16 v8h;
typedef __attribute__((ext_vector_type(8)))  float    v8f;

__device__ __forceinline__ v8f zero_v8f() {
  v8f z;
#pragma unroll
  for (int i = 0; i < 8; ++i) z[i] = 0.0f;
  return z;
}

__device__ __forceinline__ v8f wmma_f16(v16h a, v16h b, v8f c) {
  // D = A(16x32 f16) * B(32x16 f16) + C(16x16 f32)
  return __builtin_amdgcn_wmma_f32_16x16x32_f16(false, a, false, b, (short)0, c,
                                                false, false);
}

// A fragment (16x32 f16) from row-major [rows][ld].
// ISA layout: lanes 0-15 (M=lane): halves 0..7 = K 0..7, halves 8..15 = K 16..23
//             lanes 16-31 (M=lane-16): halves 0..7 = K 8..15, halves 8..15 = K 24..31
__device__ __forceinline__ v16h load_a_frag(const _Float16* __restrict__ base,
                                            int ld, int r0, int k0) {
  int lane = threadIdx.x & 31;
  int m    = lane & 15;
  int hk   = (lane >> 4) << 3;
  const _Float16* p = base + (size_t)(r0 + m) * ld + k0 + hk;
  union { v16h v; v8h h[2]; } u;
  u.h[0] = *(const v8h*)(p);
  u.h[1] = *(const v8h*)(p + 16);
  return u.v;
}

// B fragment (32x16 f16): column n of B = row n of BT ([N][K] row-major).
// ISA layout: lanes 0-15 (N=lane): halves 0..7 = K 0..7, 8..15 = K 16..23 ; lanes 16-31: K 8..15, K 24..31
__device__ __forceinline__ v16h load_b_frag(const _Float16* __restrict__ bt,
                                            int ld, int n0, int k0) {
  int lane = threadIdx.x & 31;
  int n    = n0 + (lane & 15);
  int kk   = k0 + ((lane >> 4) << 3);
  const _Float16* p = bt + (size_t)n * ld + kk;
  union { v16h v; v8h h[2]; } u;
  u.h[0] = *(const v8h*)(p);
  u.h[1] = *(const v8h*)(p + 16);
  return u.v;
}

// One wave computes a 16x64 tile of A[rows][K] @ BT^T  (BT is [N][K] row-major).
__device__ __forceinline__ void gemm_16x64(const _Float16* __restrict__ A, int lda,
                                           const _Float16* __restrict__ BT, int ldb,
                                           int r0, int n0, int K, v8f acc[4]) {
#pragma unroll
  for (int j = 0; j < 4; ++j) acc[j] = zero_v8f();
  for (int k0 = 0; k0 < K; k0 += 32) {
    v16h a = load_a_frag(A, lda, r0, k0);
#pragma unroll
    for (int j = 0; j < 4; ++j) {
      v16h b = load_b_frag(BT, ldb, n0 + 16 * j, k0);
      acc[j] = wmma_f16(a, b, acc[j]);
    }
  }
}

// ---------------- conversion kernels ----------------

__global__ __launch_bounds__(256) void cvt_f32_f16_kernel(
    const float* __restrict__ src, _Float16* __restrict__ dst, int n) {
  int i = blockIdx.x * 256 + threadIdx.x;
  if (i < n) dst[i] = (_Float16)src[i];
}

// WT[n][k] = (f16) W[k][n] ;  W is D_ x D_
__global__ __launch_bounds__(256) void transpose_cvt_kernel(
    const float* __restrict__ W, _Float16* __restrict__ WT) {
  int idx = blockIdx.x * 256 + threadIdx.x;
  if (idx >= D_ * D_) return;
  int no = idx % D_;
  int k  = idx / D_;
  WT[(size_t)no * D_ + k] = (_Float16)W[(size_t)k * D_ + no];
}

// ---------------- QKV projection + RoPE + RMS + v-mix ----------------
// grid: (BT_/64, H_, 3)  block: 128 (4 waves, each 16 rows)
__global__ __launch_bounds__(128) void qkv_kernel(
    const _Float16* __restrict__ xh,
    const _Float16* __restrict__ WqT, const _Float16* __restrict__ WkT,
    const _Float16* __restrict__ WvT,
    const float* __restrict__ cosp, const float* __restrict__ sinp,
    const float* __restrict__ v1,
    const float* __restrict__ lamb1p, const float* __restrict__ lamb2p,
    _Float16* __restrict__ qh, _Float16* __restrict__ kh,
    _Float16* __restrict__ vTh) {
  int wave = threadIdx.x >> 5;
  int lane = threadIdx.x & 31;
  int hi = lane >> 4, ln = lane & 15;
  int r0 = blockIdx.x * 64 + wave * 16;   // row in [0, BT_)
  int h  = blockIdx.y;
  int n0 = h * HD_;
  int mode = blockIdx.z;                  // 0=q 1=k 2=v

  const _Float16* BT = (mode == 0) ? WqT : (mode == 1) ? WkT : WvT;
  v8f acc[4];
  gemm_16x64(xh, D_, BT, D_, r0, n0, D_, acc);

  int bidx  = r0 >> 11;       // batch
  int tbase = r0 & (T_ - 1);  // token base (16-row strip stays in one batch)

  if (mode <= 1) {
    _Float16* dstbase = (mode == 0) ? qh : kh;
#pragma unroll
    for (int r = 0; r < 8; ++r) {
      int t = tbase + 8 * hi + r;
      const float* cr = cosp + (size_t)t * (HD_ / 2);
      const float* sr = sinp + (size_t)t * (HD_ / 2);
      float y[4];
      float ss = 0.0f;
#pragma unroll
      for (int j = 0; j < 2; ++j) {
        float c  = cr[ln + 16 * j];
        float s  = sr[ln + 16 * j];
        float x1 = acc[j][r];       // col i      (i < 32)
        float x2 = acc[j + 2][r];   // col i + 32
        float y1 = x1 * c + x2 * s;
        float y2 = x2 * c - x1 * s;
        y[j] = y1; y[j + 2] = y2;
        ss += y1 * y1 + y2 * y2;
      }
      // 16-lane butterfly: full row sum-of-squares (RoPE preserves it anyway)
      ss += __shfl_xor(ss, 1);
      ss += __shfl_xor(ss, 2);
      ss += __shfl_xor(ss, 4);
      ss += __shfl_xor(ss, 8);
      float inv = rsqrtf(ss * (1.0f / 64.0f) + 1e-8f);
      _Float16* dst = dstbase + (((size_t)bidx * H_ + h) * T_ + t) * HD_;
#pragma unroll
      for (int j = 0; j < 4; ++j) dst[ln + 16 * j] = (_Float16)(y[j] * inv);
    }
  } else {
    float l1 = lamb1p[0], l2 = lamb2p[0];
#pragma unroll
    for (int r = 0; r < 8; ++r) {
      int t = tbase + 8 * hi + r;
      const float* v1row = v1 + (((size_t)bidx * H_ + h) * T_ + t) * HD_;
#pragma unroll
      for (int j = 0; j < 4; ++j) {
        int n = ln + 16 * j;
        float val = l1 * acc[j][r] + l2 * v1row[n];
        // transposed store: vT[b,h,d,t]  -> contiguous-in-t for PV B-fragments
        vTh[(((size_t)bidx * H_ + h) * HD_ + n) * T_ + t] = (_Float16)val;
      }
    }
  }
}

// ---------------- causal flash attention ----------------
// grid: (T_/64, B_*H_)  block: 128 (4 independent waves, 16 query rows each)
__global__ __launch_bounds__(128) void attn_kernel(
    const _Float16* __restrict__ qh, const _Float16* __restrict__ kh,
    const _Float16* __restrict__ vTh, _Float16* __restrict__ yh) {
  __shared__ __align__(16) _Float16 Plds[4][16 * 32];

  int wave = threadIdx.x >> 5;
  int lane = threadIdx.x & 31;
  int hi = lane >> 4, ln = lane & 15;
  int bh   = blockIdx.y;
  int bidx = bh / H_, h = bh % H_;
  int qi0  = blockIdx.x * 64 + wave * 16;

  const _Float16* Q = qh  + (size_t)bh * T_ * HD_;
  const _Float16* K = kh  + (size_t)bh * T_ * HD_;
  const _Float16* V = vTh + (size_t)bh * HD_ * T_;  // [HD_][T_]

  v16h aq0 = load_a_frag(Q, HD_, qi0, 0);
  v16h aq1 = load_a_frag(Q, HD_, qi0, 32);

  v8f o[4];
#pragma unroll
  for (int j = 0; j < 4; ++j) o[j] = zero_v8f();
  float mrow[8], lrow[8];
#pragma unroll
  for (int r = 0; r < 8; ++r) { mrow[r] = -1e30f; lrow[r] = 0.0f; }

  _Float16* Pw = &Plds[wave][0];
  const float scale = 0.125f;  // 1/sqrt(64)

  int nblocks = (qi0 + 47) >> 5;  // keys 0 .. qi0+15 in blocks of 32
  for (int kb = 0; kb < nblocks; ++kb) {
    int kb0 = kb * 32;
    // ---- S = Q K^T  (16 x 32) ----
    v8f s[2];
#pragma unroll
    for (int j = 0; j < 2; ++j) {
      s[j] = zero_v8f();
      v16h bk0 = load_b_frag(K, HD_, kb0 + 16 * j, 0);
      s[j] = wmma_f16(aq0, bk0, s[j]);
      v16h bk1 = load_b_frag(K, HD_, kb0 + 16 * j, 32);
      s[j] = wmma_f16(aq1, bk1, s[j]);
    }
    bool edge = (kb0 + 31 > qi0);
#pragma unroll
    for (int j = 0; j < 2; ++j)
#pragma unroll
      for (int r = 0; r < 8; ++r) {
        float v = s[j][r] * scale;
        int m = 8 * hi + r;
        int n = 16 * j + ln;
        if (edge && (kb0 + n > qi0 + m)) v = -1e30f;
        s[j][r] = v;
      }
    // ---- online softmax (row stats via 16-lane butterflies) ----
#pragma unroll
    for (int r = 0; r < 8; ++r) {
      float mx = fmaxf(s[0][r], s[1][r]);
      mx = fmaxf(mx, __shfl_xor(mx, 1));
      mx = fmaxf(mx, __shfl_xor(mx, 2));
      mx = fmaxf(mx, __shfl_xor(mx, 4));
      mx = fmaxf(mx, __shfl_xor(mx, 8));
      float mnew  = fmaxf(mrow[r], mx);
      float alpha = __expf(mrow[r] - mnew);
      float p0 = __expf(s[0][r] - mnew);
      float p1 = __expf(s[1][r] - mnew);
      float rs = p0 + p1;
      rs += __shfl_xor(rs, 1);
      rs += __shfl_xor(rs, 2);
      rs += __shfl_xor(rs, 4);
      rs += __shfl_xor(rs, 8);
      lrow[r] = lrow[r] * alpha + rs;
      mrow[r] = mnew;
#pragma unroll
      for (int j2 = 0; j2 < 4; ++j2) o[j2][r] *= alpha;
      s[0][r] = p0;
      s[1][r] = p1;
    }
    // ---- P (C layout) -> LDS -> A fragment ----
#pragma unroll
    for (int r = 0; r < 8; ++r)
#pragma unroll
      for (int j = 0; j < 2; ++j)
        Pw[(8 * hi + r) * 32 + 16 * j + ln] = (_Float16)s[j][r];
    asm volatile("s_wait_dscnt 0" ::: "memory");
    union { v16h v; v8h hh[2]; } ua;
    const _Float16* pr = Pw + ln * 32 + hi * 8;
    ua.hh[0] = *(const v8h*)(pr);
    ua.hh[1] = *(const v8h*)(pr + 16);
    // ---- O += P V ----
#pragma unroll
    for (int j2 = 0; j2 < 4; ++j2) {
      v16h bv = load_b_frag(V, T_, 16 * j2, kb0);
      o[j2] = wmma_f16(ua.v, bv, o[j2]);
    }
  }

  // ---- normalize + store to y[b, t, h*64 + d] ----
#pragma unroll
  for (int r = 0; r < 8; ++r) {
    float inv = 1.0f / lrow[r];
    int t = qi0 + 8 * hi + r;
    _Float16* dst = yh + ((size_t)bidx * T_ + t) * D_ + h * HD_;
#pragma unroll
    for (int j2 = 0; j2 < 4; ++j2)
      dst[16 * j2 + ln] = (_Float16)(o[j2][r] * inv);
  }
}

// ---------------- output projection: out = y @ Wo ----------------
// grid: (BT_/64, D_/64)  block: 128
__global__ __launch_bounds__(128) void out_kernel(
    const _Float16* __restrict__ yh, const _Float16* __restrict__ WoT,
    float* __restrict__ out) {
  int wave = threadIdx.x >> 5;
  int lane = threadIdx.x & 31;
  int hi = lane >> 4, ln = lane & 15;
  int r0 = blockIdx.x * 64 + wave * 16;
  int n0 = blockIdx.y * 64;

  v8f acc[4];
  gemm_16x64(yh, D_, WoT, D_, r0, n0, D_, acc);

#pragma unroll
  for (int r = 0; r < 8; ++r) {
    int row = r0 + 8 * hi + r;
    float* dst = out + (size_t)row * D_ + n0;
#pragma unroll
    for (int j = 0; j < 4; ++j) dst[16 * j + ln] = acc[j][r];
  }
}

// ---------------- launcher ----------------
extern "C" void kernel_launch(void* const* d_in, const int* in_sizes, int n_in,
                              void* d_out, int out_size, void* d_ws, size_t ws_size,
                              hipStream_t stream) {
  (void)in_sizes; (void)n_in; (void)out_size; (void)ws_size;
  const float* x     = (const float*)d_in[0];
  const float* cosp  = (const float*)d_in[1];
  const float* sinp  = (const float*)d_in[2];
  const float* v1    = (const float*)d_in[3];
  const float* Wq    = (const float*)d_in[4];
  const float* Wk    = (const float*)d_in[5];
  const float* Wv    = (const float*)d_in[6];
  const float* Wo    = (const float*)d_in[7];
  const float* lamb1 = (const float*)d_in[8];
  const float* lamb2 = (const float*)d_in[9];
  float* out = (float*)d_out;

  char* w = (char*)d_ws;
  const size_t NBTD = (size_t)BT_ * D_;          // 6291456
  const size_t NW   = (size_t)D_ * D_;           // 589824
  _Float16* xh  = (_Float16*)w; w += NBTD * 2;
  _Float16* qh  = (_Float16*)w; w += NBTD * 2;
  _Float16* kh  = (_Float16*)w; w += NBTD * 2;
  _Float16* vTh = (_Float16*)w; w += NBTD * 2;
  _Float16* yh  = (_Float16*)w; w += NBTD * 2;
  _Float16* WqT = (_Float16*)w; w += NW * 2;
  _Float16* WkT = (_Float16*)w; w += NW * 2;
  _Float16* WvT = (_Float16*)w; w += NW * 2;
  _Float16* WoT = (_Float16*)w; w += NW * 2;

  // 1) convert activations and weights to f16 (weights transposed)
  cvt_f32_f16_kernel<<<dim3((unsigned)((NBTD + 255) / 256)), dim3(256), 0, stream>>>(
      x, xh, (int)NBTD);
  unsigned tgrid = (unsigned)((NW + 255) / 256);
  transpose_cvt_kernel<<<dim3(tgrid), dim3(256), 0, stream>>>(Wq, WqT);
  transpose_cvt_kernel<<<dim3(tgrid), dim3(256), 0, stream>>>(Wk, WkT);
  transpose_cvt_kernel<<<dim3(tgrid), dim3(256), 0, stream>>>(Wv, WvT);
  transpose_cvt_kernel<<<dim3(tgrid), dim3(256), 0, stream>>>(Wo, WoT);

  // 2) q/k/v projections + RoPE + RMS-norm + v-mix
  qkv_kernel<<<dim3(BT_ / 64, H_, 3), dim3(128), 0, stream>>>(
      xh, WqT, WkT, WvT, cosp, sinp, v1, lamb1, lamb2, qh, kh, vTh);

  // 3) causal flash attention
  attn_kernel<<<dim3(T_ / 64, B_ * H_), dim3(128), 0, stream>>>(qh, kh, vTh, yh);

  // 4) output projection
  out_kernel<<<dim3(BT_ / 64, D_ / 64), dim3(128), 0, stream>>>(yh, WoT, out);
}
